// HT_43284680409622
// MI455X (gfx1250) — compile-verified, matrix operand order checked by reading, not run on hardware
//
#include <hip/hip_runtime.h>
#include <hip/hip_bf16.h>
#include <math.h>

// ---------------------------------------------------------------------------
// MI455X / gfx1250 implementation (miss_ent_domain == 1 path).
//
//   norm      : e_all_h = f16(BN(E_weight));  ef[s]=BN(E[e_idx[s]]); rf=BN(R[r_idx])
//   build P   : P[b,(i,j)] = e4*e3                  (256 x 65536 f16)
//   GEMM1     : rvecT[a,b] = HRh[a,:] . P[b,:]      (128 x 256, K=65536, split-K 64)
//   build Q   : Q[b,(d,j)] = r*rvec                 (256 x 16384 f16)
//   GEMM2     : w[b,i]    = Q[b,:] . rootT[i,:]     (256 x 128, K=16384, split-K 32)
//   build G   : G[b,(a,i)] = w*e2                   (256 x 32768 f16)
//   GEMM3     : v[b,j]    = G[b,:] . hlT[j,:]       (256 x 256, K=32768, split-K 32)
//   GEMM4     : scores[b,n] = vh[b,:] . e_all_h[n,:]  (256 x 100000, K=256)
//   softmax   : row-wise over 100000
//
// v_wmma_f32_16x16x32_f16 (f32 WMMA is only 16x16x4 -> 8x less work/instr;
// BN-normalized data makes f16-in/f32-acc the right precision).  Waves
// register-block MRxNR 16x16 tiles; split-K spreads the small-output GEMMs
// across WGPs; fixed-order reduce keeps results bitwise deterministic.
// Wave index goes through readfirstlane (scalar loop control, EXEC untouched
// around WMMA); K-loop is unrolled x4 with per-lane fragment pointers bumped
// once per body (unrolled loads use immediate offsets).  Fragment pointers
// are explicitly addrspace(1) so loads emit global_load_b128 (LOADcnt only),
// not flat_load_b128 (which also burns DScnt / the LDS path, ISA 10.1).
// ---------------------------------------------------------------------------

typedef __attribute__((ext_vector_type(16))) _Float16 v16h;
typedef __attribute__((ext_vector_type(8)))  _Float16 v8h;
typedef __attribute__((ext_vector_type(8)))  float    v8f;

#define GAS __attribute__((address_space(1)))   // AMDGPU global address space

#define N_ENT 100000
#define N_REL 200
#define DE    256
#define DR    128
#define RANK  128
#define BATCH 256
#define EPS   1e-5f

// ---- WMMA fragment helpers (CDNA5 ISA 7.12.2 layouts, wave32) --------------
// A fragment 16x32 f16 from row-major-over-K storage:
//   lane<16 : M=lane,    halfs 0..7 = K k0..+7,  halfs 8..15 = K k0+16..+23
//   lane>=16: M=lane-16, halfs 0..7 = K k0+8..+15, halfs 8..15 = K k0+24..+31
// B (32x16) mirrors A when stored transposed (N x K row-major).
// p is the per-lane pointer  X + (row0 + m)*ld + k0 + kb,  m=lane&15,
// kb = 8*(lane>=16); the two 16B loads then sit at p and p+16.
__device__ __forceinline__ v16h load_frag16p(GAS const _Float16* p) {
  v8h lo = *(GAS const v8h*)(p);
  v8h hi = *(GAS const v8h*)(p + 16);
  v16h f;
#pragma unroll
  for (int t = 0; t < 8; ++t) { f[t] = lo[t]; f[8 + t] = hi[t]; }
  return f;
}

// D/C 16x16 f32: VGPR r -> row (r + 8*(lane>=16)), col = lane&15.
__device__ __forceinline__ void store_frag_f32(GAS float* C, size_t ld,
                                               int row0, size_t col0, v8f d) {
  const int lane = threadIdx.x & 31;
  const int n    = lane & 15;
  const int mb   = (lane >> 4) << 3;
#pragma unroll
  for (int r = 0; r < 8; ++r)
    C[(size_t)(row0 + mb + r) * ld + col0 + (size_t)n] = d[r];
}

// ---- register-blocked split-K f16 GEMM ------------------------------------
// Cpart[s][M][N] = A[M,K] * Bt[N,K]^T restricted to K-slab s.
// Requires M % (16*MR) == 0, N % (16*NR) == 0, K == S*Kc, Kc % 128 == 0.
template <int MR, int NR>
__global__ void ht_gemm_f16(const _Float16* __restrict__ A,
                            const _Float16* __restrict__ Bt,
                            float* __restrict__ Cpart,
                            int M, int N, size_t K, int S, size_t Kc) {
  const int wavesPerBlk = blockDim.x >> 5;
  // readfirstlane: provably wave-uniform -> scalar loop control.
  const int waveInBlk = __builtin_amdgcn_readfirstlane((int)(threadIdx.x >> 5));
  const int wave   = blockIdx.x * wavesPerBlk + waveInBlk;
  const int nWaves = gridDim.x * wavesPerBlk;
  const int Mtb = M / (16 * MR), Ntb = N / (16 * NR);
  const int tilesPerSlab = Mtb * Ntb;
  const int total = tilesPerSlab * S;

  const int lane = threadIdx.x & 31;
  const int m    = lane & 15;
  const int kb   = (lane >> 4) << 3;            // 0 or 8

  for (int t = wave; t < total; t += nWaves) {
    const int s   = t / tilesPerSlab;
    const int rem = t - s * tilesPerSlab;
    const int mtb = rem / Ntb;
    const int ntb = rem - mtb * Ntb;
    const int mrow = mtb * 16 * MR;
    const int ncol = ntb * 16 * NR;
    const size_t kBeg = (size_t)s * Kc;

    // per-lane fragment stream pointers in explicit global AS
    GAS const _Float16* pa[MR];
    GAS const _Float16* pb[NR];
#pragma unroll
    for (int i = 0; i < MR; ++i)
      pa[i] = (GAS const _Float16*)(A + (size_t)(mrow + 16 * i + m) * K + kBeg + (size_t)kb);
#pragma unroll
    for (int j = 0; j < NR; ++j)
      pb[j] = (GAS const _Float16*)(Bt + (size_t)(ncol + 16 * j + m) * K + kBeg + (size_t)kb);

    v8f acc[MR][NR];
#pragma unroll
    for (int i = 0; i < MR; ++i)
#pragma unroll
      for (int j = 0; j < NR; ++j)
        acc[i][j] = (v8f){0.f, 0.f, 0.f, 0.f, 0.f, 0.f, 0.f, 0.f};

    for (size_t kIt = 0; kIt < Kc; kIt += 128) {   // x4-unrolled body
      // speculative per-lane prefetch: each lane prefetches its own row's
      // future line (ISA 10.5: translation misses silently dropped).
#pragma unroll
      for (int i = 0; i < MR; ++i)
        __builtin_prefetch((const _Float16*)pa[i] + 512, 0, 1);
#pragma unroll
      for (int j = 0; j < NR; ++j)
        __builtin_prefetch((const _Float16*)pb[j] + 512, 0, 1);

#pragma unroll
      for (int u = 0; u < 4; ++u) {                // offsets become immediates
        v16h af[MR], bf[NR];
#pragma unroll
        for (int i = 0; i < MR; ++i) af[i] = load_frag16p(pa[i] + u * 32);
#pragma unroll
        for (int j = 0; j < NR; ++j) bf[j] = load_frag16p(pb[j] + u * 32);
#pragma unroll
        for (int i = 0; i < MR; ++i)
#pragma unroll
          for (int j = 0; j < NR; ++j)
            acc[i][j] = __builtin_amdgcn_wmma_f32_16x16x32_f16(
                false, af[i], false, bf[j], (short)0, acc[i][j], false, false);
      }
#pragma unroll
      for (int i = 0; i < MR; ++i) pa[i] += 128;
#pragma unroll
      for (int j = 0; j < NR; ++j) pb[j] += 128;
    }

    GAS float* Cs = (GAS float*)(Cpart + (size_t)s * M * N);
#pragma unroll
    for (int i = 0; i < MR; ++i)
#pragma unroll
      for (int j = 0; j < NR; ++j)
        store_frag_f32(Cs, (size_t)N, mrow + 16 * i, (size_t)(ncol + 16 * j),
                       acc[i][j]);
  }
}

// fixed-order split-K reduce (deterministic)
__global__ void k_reduce(const float* __restrict__ Cpart, float* __restrict__ C,
                         int S, size_t MN) {
  size_t idx = (size_t)blockIdx.x * blockDim.x + threadIdx.x;
  if (idx >= MN) return;
  float s = 0.f;
  for (int i = 0; i < S; ++i) s += Cpart[(size_t)i * MN + idx];
  C[idx] = s;
}

// ---- elementwise / prep kernels -------------------------------------------
__device__ __forceinline__ float bn1(float x, float g, float b, float m, float v) {
  return (x - m) * rsqrtf(v + EPS) * g + b;
}

__global__ void k_norm_eall(const float* __restrict__ E,
                            const float* __restrict__ g, const float* __restrict__ b,
                            const float* __restrict__ m, const float* __restrict__ v,
                            _Float16* __restrict__ out) {
  size_t idx = (size_t)blockIdx.x * blockDim.x + threadIdx.x;
  if (idx >= (size_t)N_ENT * DE) return;
  int j = (int)(idx & (DE - 1));
  out[idx] = (_Float16)bn1(E[idx], g[j], b[j], m[j], v[j]);
}

__global__ void k_gather_e(const float* __restrict__ E, const int* __restrict__ eidx,
                           const float* __restrict__ g, const float* __restrict__ b,
                           const float* __restrict__ m, const float* __restrict__ v,
                           float* __restrict__ ef) {     // [3][BATCH][DE] f32
  int idx = blockIdx.x * blockDim.x + threadIdx.x;
  if (idx >= 3 * BATCH * DE) return;
  int j  = idx & (DE - 1);
  int sb = idx >> 8;
  int row = eidx[sb];
  ef[idx] = bn1(E[(size_t)row * DE + j], g[j], b[j], m[j], v[j]);
}

__global__ void k_norm_r(const float* __restrict__ R, const int* __restrict__ ridx,
                         const float* __restrict__ g, const float* __restrict__ b,
                         const float* __restrict__ m, const float* __restrict__ v,
                         float* __restrict__ rf) {       // [BATCH][DR] f32
  int idx = blockIdx.x * blockDim.x + threadIdx.x;
  if (idx >= BATCH * DR) return;
  int d  = idx & (DR - 1);
  int bb = idx >> 7;
  rf[idx] = bn1(R[(size_t)ridx[bb] * DR + d], g[d], b[d], m[d], v[d]);
}

__global__ void k_conv_f16(const float* __restrict__ in, _Float16* __restrict__ out,
                           size_t n) {
  size_t idx = (size_t)blockIdx.x * blockDim.x + threadIdx.x;
  if (idx < n) out[idx] = (_Float16)in[idx];
}

// hlT[j, a*256+i] = ht_left[a,i,j]  -- LDS-tiled 32x32 transpose:
// coalesced reads (contiguous j), coalesced writes (contiguous i).
__global__ void k_conv_hlT(const float* __restrict__ hl, _Float16* __restrict__ out) {
  __shared__ float tile[32][33];                 // pad -> no bank conflicts
  const int bid = blockIdx.x;                    // 128 * 8 * 8 tiles
  const int a   = bid >> 6;
  const int rem = bid & 63;
  const int ti  = (rem >> 3) << 5;               // i tile origin
  const int tj  = (rem & 7) << 5;                // j tile origin
  const int tx  = threadIdx.x & 31;
  const int ty  = threadIdx.x >> 5;              // 0..7
  const float* src = hl + (size_t)a * DE * DE;
#pragma unroll
  for (int k = 0; k < 4; ++k) {
    int r = ty + k * 8;
    tile[r][tx] = src[(size_t)(ti + r) * DE + tj + tx];
  }
  __syncthreads();
#pragma unroll
  for (int k = 0; k < 4; ++k) {
    int r = ty + k * 8;                          // j within tile
    out[(size_t)(tj + r) * ((size_t)RANK * DE) + (size_t)a * DE + ti + tx] =
        (_Float16)tile[tx][r];
  }
}

// rootT[i, d*128+j] = ht_root[d,i,j]  (inner j contiguous on both sides)
__global__ void k_conv_rootT(const float* __restrict__ rt, _Float16* __restrict__ out) {
  size_t idx = (size_t)blockIdx.x * blockDim.x + threadIdx.x;
  if (idx >= (size_t)DR * RANK * RANK) return;
  int i   = (int)(idx >> 14);
  int rem = (int)(idx & 16383);
  int d   = rem >> 7;
  int j   = rem & 127;
  out[idx] = (_Float16)rt[(size_t)d * RANK * RANK + (size_t)i * RANK + j];
}

// P[b, i*256+j] = e4[b,i] * e3[b,j]
__global__ void k_build_P(const float* __restrict__ ef, _Float16* __restrict__ P) {
  size_t idx = (size_t)blockIdx.x * blockDim.x + threadIdx.x;
  if (idx >= (size_t)BATCH * DE * DE) return;
  int b = (int)(idx >> 16);
  int k = (int)(idx & 65535);
  int i = k >> 8, j = k & 255;
  const float* e3 = ef + (size_t)1 * BATCH * DE;
  const float* e4 = ef + (size_t)2 * BATCH * DE;
  P[idx] = (_Float16)(e4[b * DE + i] * e3[b * DE + j]);
}

// Q[b, d*128+j] = r[b,d] * rvec[b,j],  rvec[b,j] = Crv[j*256 + b]
__global__ void k_build_Q(const float* __restrict__ rf, const float* __restrict__ Crv,
                          _Float16* __restrict__ Q) {
  size_t idx = (size_t)blockIdx.x * blockDim.x + threadIdx.x;
  if (idx >= (size_t)BATCH * DR * RANK) return;
  int b = (int)(idx >> 14);
  int k = (int)(idx & 16383);
  int d = k >> 7, j = k & 127;
  Q[idx] = (_Float16)(rf[b * DR + d] * Crv[(size_t)j * BATCH + b]);
}

// G[b, a*256+i] = w[b,a] * e2[b,i]
__global__ void k_build_G(const float* __restrict__ Cw, const float* __restrict__ ef,
                          _Float16* __restrict__ G) {
  size_t idx = (size_t)blockIdx.x * blockDim.x + threadIdx.x;
  if (idx >= (size_t)BATCH * RANK * DE) return;
  int b = (int)(idx >> 15);
  int k = (int)(idx & 32767);
  int a = k >> 8, i = k & 255;
  G[idx] = (_Float16)(Cw[b * RANK + a] * ef[b * DE + i]);
}

// ---- row softmax over N_ENT ------------------------------------------------
__global__ void k_softmax(float* __restrict__ S) {
  __shared__ float red[256];
  const int b = blockIdx.x;
  const int t = threadIdx.x;
  float* row = S + (size_t)b * N_ENT;

  float mx = -3.4e38f;
  for (int n = t; n < N_ENT; n += 256) mx = fmaxf(mx, row[n]);
  red[t] = mx; __syncthreads();
  for (int s = 128; s > 0; s >>= 1) {
    if (t < s) red[t] = fmaxf(red[t], red[t + s]);
    __syncthreads();
  }
  mx = red[0]; __syncthreads();

  float sum = 0.f;
  for (int n = t; n < N_ENT; n += 256) sum += expf(row[n] - mx);
  red[t] = sum; __syncthreads();
  for (int s = 128; s > 0; s >>= 1) {
    if (t < s) red[t] += red[t + s];
    __syncthreads();
  }
  const float inv = 1.0f / red[0];
  __syncthreads();

  for (int n = t; n < N_ENT; n += 256) row[n] = expf(row[n] - mx) * inv;
}

// ---------------------------------------------------------------------------
static inline size_t alignUp(size_t x) { return (x + 255) & ~(size_t)255; }

extern "C" void kernel_launch(void* const* d_in, const int* in_sizes, int n_in,
                              void* d_out, int out_size, void* d_ws, size_t ws_size,
                              hipStream_t stream) {
  const int*   r_idx  = (const int*)  d_in[0];
  const int*   e_idx  = (const int*)  d_in[1];
  const float* E_w    = (const float*)d_in[2];
  const float* R_w    = (const float*)d_in[3];
  const float* ht_root= (const float*)d_in[4];
  const float* ht_left= (const float*)d_in[5];
  const float* ht_rght= (const float*)d_in[6];
  const float* bne_g  = (const float*)d_in[7];
  const float* bne_b  = (const float*)d_in[8];
  const float* bne_m  = (const float*)d_in[9];
  const float* bne_v  = (const float*)d_in[10];
  const float* bnr_g  = (const float*)d_in[11];
  const float* bnr_b  = (const float*)d_in[12];
  const float* bnr_m  = (const float*)d_in[13];
  const float* bnr_v  = (const float*)d_in[14];
  // d_in[15] = miss_ent_domain (harness provides 1; path specialized)

  float* scores = (float*)d_out;                 // [BATCH][N_ENT] f32

  // ---- workspace carve ----
  char* p = (char*)d_ws;
  _Float16* e_all_h = (_Float16*)p; p += alignUp((size_t)N_ENT * DE * 2);
  _Float16* HRh     = (_Float16*)p; p += alignUp((size_t)RANK * DE * DE * 2);
  _Float16* hlT     = (_Float16*)p; p += alignUp((size_t)RANK * DE * DE * 2);
  _Float16* rootT   = (_Float16*)p; p += alignUp((size_t)DR * RANK * RANK * 2);
  float*    ef      = (float*)p;    p += alignUp((size_t)3 * BATCH * DE * 4);
  float*    rf      = (float*)p;    p += alignUp((size_t)BATCH * DR * 4);
  _Float16* P       = (_Float16*)p; p += alignUp((size_t)BATCH * DE * DE * 2);
  float*    Crv     = (float*)p;    p += alignUp((size_t)RANK * BATCH * 4);
  _Float16* Q       = (_Float16*)p; p += alignUp((size_t)BATCH * DR * RANK * 2);
  float*    Cw      = (float*)p;    p += alignUp((size_t)BATCH * RANK * 4);
  _Float16* G       = (_Float16*)p; p += alignUp((size_t)BATCH * RANK * DE * 2);
  float*    Cv      = (float*)p;    p += alignUp((size_t)BATCH * DE * 4);
  _Float16* vh      = (_Float16*)p; p += alignUp((size_t)BATCH * DE * 2);
  float*    Cpart   = (float*)p;    p += alignUp((size_t)64 * RANK * BATCH * 4); // 8.4MB, reused
  (void)ws_size; (void)n_in; (void)in_sizes; (void)out_size;

  const int T = 256;

  // normalize + precision staging
  k_norm_eall<<<(N_ENT * DE + T - 1) / T, T, 0, stream>>>(E_w, bne_g, bne_b, bne_m, bne_v, e_all_h);
  k_gather_e <<<(3 * BATCH * DE) / T, T, 0, stream>>>(E_w, e_idx, bne_g, bne_b, bne_m, bne_v, ef);
  k_norm_r   <<<(BATCH * DR) / T, T, 0, stream>>>(R_w, r_idx, bnr_g, bnr_b, bnr_m, bnr_v, rf);
  k_conv_f16 <<<((size_t)RANK * DE * DE + T - 1) / T, T, 0, stream>>>(ht_rght, HRh, (size_t)RANK * DE * DE);
  k_conv_hlT <<<RANK * 8 * 8, T, 0, stream>>>(ht_left, hlT);
  k_conv_rootT<<<((size_t)DR * RANK * RANK + T - 1) / T, T, 0, stream>>>(ht_root, rootT);

  // stage 1: rvecT[a,b]   M=128 N=256 K=65536, split-K 64 -> 1024 wave-tiles
  k_build_P<<<((size_t)BATCH * DE * DE + T - 1) / T, T, 0, stream>>>(ef, P);
  ht_gemm_f16<2, 4><<<128, T, 0, stream>>>(HRh, P, Cpart, RANK, BATCH,
                                           (size_t)DE * DE, 64, 1024);
  k_reduce<<<(RANK * BATCH + T - 1) / T, T, 0, stream>>>(Cpart, Crv, 64, (size_t)RANK * BATCH);

  // stage 2: w[b,i]       M=256 N=128 K=16384, split-K 32 -> 512 wave-tiles
  k_build_Q<<<((size_t)BATCH * DR * RANK + T - 1) / T, T, 0, stream>>>(rf, Crv, Q);
  ht_gemm_f16<2, 4><<<64, T, 0, stream>>>(Q, rootT, Cpart, BATCH, RANK,
                                          (size_t)DR * RANK, 32, 512);
  k_reduce<<<(BATCH * RANK + T - 1) / T, T, 0, stream>>>(Cpart, Cw, 32, (size_t)BATCH * RANK);

  // stage 3: v[b,j]       M=256 N=256 K=32768, split-K 32 -> 1024 wave-tiles
  k_build_G<<<((size_t)BATCH * RANK * DE + T - 1) / T, T, 0, stream>>>(Cw, ef, G);
  ht_gemm_f16<2, 4><<<128, T, 0, stream>>>(G, hlT, Cpart, BATCH, DE,
                                           (size_t)RANK * DE, 32, 1024);
  k_reduce<<<(BATCH * DE + T - 1) / T, T, 0, stream>>>(Cpart, Cv, 32, (size_t)BATCH * DE);

  // stage 4: scores       M=256 N=100000 K=256  (3125 N-blocks of 32)
  k_conv_f16<<<(BATCH * DE) / T, T, 0, stream>>>(Cv, vh, (size_t)BATCH * DE);
  ht_gemm_f16<2, 2><<<3125, T, 0, stream>>>(vh, e_all_h, scores, BATCH, N_ENT,
                                            (size_t)DE, 1, (size_t)DE);

  // stage 5: softmax rows
  k_softmax<<<BATCH, T, 0, stream>>>(scores);
}
